// Encoder_Decoder_60146722013205
// MI455X (gfx1250) — compile-verified, matrix operand order checked by reading, not run on hardware
//
#include <hip/hip_runtime.h>
#include <hip/hip_bf16.h>

// ---------------------------------------------------------------------------
// Types for CDNA5 WMMA (wave32): D(f32 16x16) = A(bf16 16x32) x B(bf16 32x16) + C
// ---------------------------------------------------------------------------
typedef __bf16 v16bf __attribute__((ext_vector_type(16)));
typedef float  v8f   __attribute__((ext_vector_type(8)));

struct alignas(16) U128 { unsigned int x, y, z, w; };
union FragU { U128 q[2]; v16bf v; };

__device__ __forceinline__ unsigned short f2bf(float f) {
    unsigned int u = __float_as_uint(f);
    unsigned int lsb = (u >> 16) & 1u;
    u += 0x7fffu + lsb;                 // round-to-nearest-even
    return (unsigned short)(u >> 16);
}
__device__ __forceinline__ unsigned int pack2(float x, float y) {
    return (unsigned int)f2bf(x) | ((unsigned int)f2bf(y) << 16);
}

// Stage one (A:128x32, B:32x64->transposed) tile into an LDS buffer as bf16.
__device__ __forceinline__ void stage_tiles(
    unsigned short* __restrict__ AsB, unsigned short* __restrict__ BsB,
    int ar, int ac, int br, int bc,
    const float4& a0, const float4& a1, const float4& a2, const float4& a3,
    const float4& b0, const float4& b1)
{
    U128 p0, p1;
    p0.x = pack2(a0.x, a0.y); p0.y = pack2(a0.z, a0.w);
    p0.z = pack2(a1.x, a1.y); p0.w = pack2(a1.z, a1.w);
    p1.x = pack2(a2.x, a2.y); p1.y = pack2(a2.z, a2.w);
    p1.z = pack2(a3.x, a3.y); p1.w = pack2(a3.z, a3.w);
    *reinterpret_cast<U128*>(&AsB[ar * 32 + ac])     = p0;
    *reinterpret_cast<U128*>(&AsB[ar * 32 + ac + 8]) = p1;

    unsigned short bt[8] = { f2bf(b0.x), f2bf(b0.y), f2bf(b0.z), f2bf(b0.w),
                             f2bf(b1.x), f2bf(b1.y), f2bf(b1.z), f2bf(b1.w) };
#pragma unroll
    for (int j = 0; j < 8; ++j) BsB[(bc + j) * 32 + br] = bt[j];
}

// ---------------------------------------------------------------------------
// GEMM: C[M,N] = act( A[M,K] @ B[K,N] + bias[N] ) (+ resid)
// bf16 WMMA math, fp32 accumulation.  flags: bit0 = ReLU, bit1 = +resid.
// Block = 256 threads (8 waves), 128x64 C tile, K stepped by 32,
// double-buffered LDS, 32x32 register blocking per wave (4 WMMAs / K-step).
// ---------------------------------------------------------------------------
__global__ __launch_bounds__(256) void gemm_bf16_wmma(
    const float* __restrict__ A, const float* __restrict__ B,
    const float* __restrict__ bias, const float* __restrict__ resid,
    float* __restrict__ C, int M, int N, int K, int flags)
{
    __shared__ __align__(16) unsigned short As[2][128 * 32];  // [m][k]
    __shared__ __align__(16) unsigned short Bs[2][64 * 32];   // [n][k] (transposed)

    const int tid  = threadIdx.x;
    const int m0   = blockIdx.y * 128;
    const int n0   = blockIdx.x * 64;
    const int w    = tid >> 5;
    const int lane = tid & 31;
    const int mW   = (w >> 1) * 32;   // 0,32,64,96
    const int nW   = (w & 1) * 32;    // 0,32

    // staging maps: A: 128 rows x 32 cols, 16 floats/thread (half a row)
    const int ar = tid >> 1;          // 0..127
    const int ac = (tid & 1) * 16;    // 0 or 16
    // B: 32 k-rows x 64 n-cols, 8 floats/thread along N
    const int br = tid >> 3;          // 0..31
    const int bc = (tid & 7) * 8;     // 0..56

    // fragment geometry (ISA 16-bit A/B layout):
    // lanes 0..15 hold K=[0..7]+[16..23]; lanes 16..31 hold K=[8..15]+[24..31]
    const int sel   = lane >> 4;
    const int mrow0 = mW + (lane & 15);
    const int mrow1 = mrow0 + 16;
    const int ncol0 = nW + (lane & 15);
    const int ncol1 = ncol0 + 16;

    v8f acc00 = {}, acc01 = {}, acc10 = {}, acc11 = {};

    const int nsteps = K >> 5;

    // ---- prologue: stage K-step 0 into buffer 0 ----
    {
        const float4* pA4 = reinterpret_cast<const float4*>(
            A + (size_t)(m0 + ar) * K + ac);
        float4 a0 = pA4[0], a1 = pA4[1], a2 = pA4[2], a3 = pA4[3];
        const float4* pB4 = reinterpret_cast<const float4*>(
            B + (size_t)br * N + (n0 + bc));
        float4 b0 = pB4[0], b1 = pB4[1];
        stage_tiles(As[0], Bs[0], ar, ac, br, bc, a0, a1, a2, a3, b0, b1);
    }
    __syncthreads();

    for (int s = 0; s < nsteps; ++s) {
        const int  cur  = s & 1;
        const bool more = (s + 1) < nsteps;

        // ---- prefetch next tile from global into registers ----
        float4 a0, a1, a2, a3, b0, b1;
        if (more) {
            const int k0 = (s + 1) << 5;
            const float4* pA4 = reinterpret_cast<const float4*>(
                A + (size_t)(m0 + ar) * K + (k0 + ac));
            a0 = pA4[0]; a1 = pA4[1]; a2 = pA4[2]; a3 = pA4[3];
            const float4* pB4 = reinterpret_cast<const float4*>(
                B + (size_t)(k0 + br) * N + (n0 + bc));
            b0 = pB4[0]; b1 = pB4[1];
        }

        // ---- fragments (aligned 16B LDS loads) + 4 WMMAs ----
        {
            const U128* r;
            FragU fa0, fa1, fb0, fb1;
            r = reinterpret_cast<const U128*>(&As[cur][mrow0 * 32]);
            fa0.q[0] = r[sel]; fa0.q[1] = r[sel + 2];
            r = reinterpret_cast<const U128*>(&As[cur][mrow1 * 32]);
            fa1.q[0] = r[sel]; fa1.q[1] = r[sel + 2];
            r = reinterpret_cast<const U128*>(&Bs[cur][ncol0 * 32]);
            fb0.q[0] = r[sel]; fb0.q[1] = r[sel + 2];
            r = reinterpret_cast<const U128*>(&Bs[cur][ncol1 * 32]);
            fb1.q[0] = r[sel]; fb1.q[1] = r[sel + 2];

            acc00 = __builtin_amdgcn_wmma_f32_16x16x32_bf16(
                        false, fa0.v, false, fb0.v, (short)0, acc00, false, false);
            acc01 = __builtin_amdgcn_wmma_f32_16x16x32_bf16(
                        false, fa0.v, false, fb1.v, (short)0, acc01, false, false);
            acc10 = __builtin_amdgcn_wmma_f32_16x16x32_bf16(
                        false, fa1.v, false, fb0.v, (short)0, acc10, false, false);
            acc11 = __builtin_amdgcn_wmma_f32_16x16x32_bf16(
                        false, fa1.v, false, fb1.v, (short)0, acc11, false, false);
        }

        // ---- store prefetched tile into the other buffer ----
        if (more) stage_tiles(As[cur ^ 1], Bs[cur ^ 1], ar, ac, br, bc,
                              a0, a1, a2, a3, b0, b1);
        __syncthreads();
    }

    // ---- epilogue: C layout — VGPR v holds M = v + 8*(lane>>4), N = lane&15 ----
    const int rb0 = m0 + mW + sel * 8;     // rows for acc0x
    const int rb1 = rb0 + 16;              // rows for acc1x
    const int gc0 = n0 + ncol0;
    const int gc1 = gc0 + 16;
    const float bi0 = bias[gc0];
    const float bi1 = bias[gc1];
#pragma unroll
    for (int vv = 0; vv < 8; ++vv) {
        float v00 = acc00[vv] + bi0, v01 = acc01[vv] + bi1;
        float v10 = acc10[vv] + bi0, v11 = acc11[vv] + bi1;
        if (flags & 1) {
            v00 = fmaxf(v00, 0.0f); v01 = fmaxf(v01, 0.0f);
            v10 = fmaxf(v10, 0.0f); v11 = fmaxf(v11, 0.0f);
        }
        const size_t o00 = (size_t)(rb0 + vv) * N + gc0;
        const size_t o01 = (size_t)(rb0 + vv) * N + gc1;
        const size_t o10 = (size_t)(rb1 + vv) * N + gc0;
        const size_t o11 = (size_t)(rb1 + vv) * N + gc1;
        if (flags & 2) {
            v00 += resid[o00]; v01 += resid[o01];
            v10 += resid[o10]; v11 += resid[o11];
        }
        C[o00] = v00; C[o01] = v01;
        C[o10] = v10; C[o11] = v11;
    }
}

// ---------------------------------------------------------------------------
// Embedding + (batch-index) positional encoding:  x[b,s,:] = emb[tok[b,s],:]+pe(b)
// Faithful to the reference bug: pe depends on batch index b only.
// ---------------------------------------------------------------------------
__global__ __launch_bounds__(256) void embed_pe(
    const int* __restrict__ tok, const float* __restrict__ emb,
    float* __restrict__ O, int L)
{
    size_t i = (size_t)blockIdx.x * 256 + threadIdx.x;   // over B*L*512
    int d = (int)(i & 511);
    size_t row = i >> 9;                                  // token row
    int b = (int)(row / L);
    int t = tok[row];
    int i2 = d & ~1;
    float ang = (float)b * __expf(-(float)i2 * (9.210340371976184f / 512.0f));
    float pe = (d & 1) ? __cosf(ang) : __sinf(ang);
    O[i] = emb[(size_t)t * 512 + d] + pe;
}

// ---------------------------------------------------------------------------
// LayerNorm over D=512 (one block per row, 2 elems/thread)
// ---------------------------------------------------------------------------
__global__ __launch_bounds__(256) void lnorm512(
    const float* __restrict__ X, const float* __restrict__ g,
    const float* __restrict__ bb, float* __restrict__ O)
{
    const int r = blockIdx.x;
    const int t = threadIdx.x;
    __shared__ float red[256];
    const float* x = X + (size_t)r * 512;
    float a = x[t], c = x[t + 256];
    red[t] = a + c;
    __syncthreads();
    for (int s = 128; s > 0; s >>= 1) { if (t < s) red[t] += red[t + s]; __syncthreads(); }
    float mean = red[0] * (1.0f / 512.0f);
    __syncthreads();
    float da = a - mean, dc = c - mean;
    red[t] = da * da + dc * dc;
    __syncthreads();
    for (int s = 128; s > 0; s >>= 1) { if (t < s) red[t] += red[t + s]; __syncthreads(); }
    float rstd = rsqrtf(red[0] * (1.0f / 512.0f) + 1e-6f);
    O[(size_t)r * 512 + t]       = da * rstd * g[t]       + bb[t];
    O[(size_t)r * 512 + t + 256] = dc * rstd * g[t + 256] + bb[t + 256];
}

// ---------------------------------------------------------------------------
// Attention: one block per (q, h, b). Q/K/V row stride 512, head offset h*64.
// mask element = msk[b*mBStride + q*mQStride + j]; 0 -> score = -1e9.
// ---------------------------------------------------------------------------
__global__ __launch_bounds__(256) void attn_soft(
    const float* __restrict__ Q, const float* __restrict__ Km,
    const float* __restrict__ Vm, const int* __restrict__ mask,
    float* __restrict__ O, int slq, int slk, int mBStride, int mQStride)
{
    const int b = blockIdx.z, h = blockIdx.y, q = blockIdx.x;
    const int t = threadIdx.x;
    __shared__ float sc[512];
    __shared__ float red[256];
    __shared__ float qv[64];

    const float* qp = Q + ((size_t)(b * slq + q)) * 512 + h * 64;
    if (t < 64) qv[t] = qp[t];
    __syncthreads();

    float lmax = -1e30f;
    for (int j = t; j < slk; j += 256) {
        const float* kp = Km + ((size_t)(b * slk + j)) * 512 + h * 64;
        float s = 0.0f;
#pragma unroll 8
        for (int d = 0; d < 64; ++d) s += qv[d] * kp[d];
        s *= 0.125f;                                  // 1/sqrt(64)
        if (mask[(size_t)b * mBStride + (size_t)q * mQStride + j] == 0) s = -1e9f;
        sc[j] = s;
        lmax = fmaxf(lmax, s);
    }
    red[t] = lmax;
    __syncthreads();
    for (int s = 128; s > 0; s >>= 1) { if (t < s) red[t] = fmaxf(red[t], red[t + s]); __syncthreads(); }
    const float mx = red[0];
    __syncthreads();

    float lsum = 0.0f;
    for (int j = t; j < slk; j += 256) {
        float e = __expf(sc[j] - mx);
        sc[j] = e;
        lsum += e;
    }
    red[t] = lsum;
    __syncthreads();
    for (int s = 128; s > 0; s >>= 1) { if (t < s) red[t] += red[t + s]; __syncthreads(); }
    const float inv = 1.0f / red[0];
    __syncthreads();

    if (t < 64) {
        float acc = 0.0f;
        for (int j = 0; j < slk; ++j)
            acc += sc[j] * Vm[((size_t)(b * slk + j)) * 512 + h * 64 + t];
        O[((size_t)(b * slq + q)) * 512 + h * 64 + t] = acc * inv;
    }
}

// ---------------------------------------------------------------------------
// In-place row log_softmax over N columns (one block per row)
// ---------------------------------------------------------------------------
__global__ __launch_bounds__(256) void log_softmax_rows(float* __restrict__ X, int N)
{
    const int r = blockIdx.x;
    const int t = threadIdx.x;
    float* x = X + (size_t)r * N;
    __shared__ float red[256];

    float lm = -1e30f;
    for (int j = t; j < N; j += 256) lm = fmaxf(lm, x[j]);
    red[t] = lm;
    __syncthreads();
    for (int s = 128; s > 0; s >>= 1) { if (t < s) red[t] = fmaxf(red[t], red[t + s]); __syncthreads(); }
    const float mx = red[0];
    __syncthreads();

    float ls = 0.0f;
    for (int j = t; j < N; j += 256) ls += __expf(x[j] - mx);
    red[t] = ls;
    __syncthreads();
    for (int s = 128; s > 0; s >>= 1) { if (t < s) red[t] += red[t + s]; __syncthreads(); }
    const float lse = mx + __logf(red[0]);
    __syncthreads();

    for (int j = t; j < N; j += 256) x[j] -= lse;
}

// ---------------------------------------------------------------------------
// Host-side orchestration
// ---------------------------------------------------------------------------
extern "C" void kernel_launch(void* const* d_in, const int* in_sizes, int n_in,
                              void* d_out, int out_size, void* d_ws, size_t ws_size,
                              hipStream_t stream)
{
    (void)in_sizes; (void)n_in; (void)out_size; (void)ws_size;

    // ---- constants ----
    const int BS = 8, LS = 512, LT = 256, D = 512, F = 2048, NL = 6, VT = 32000;
    const int Menc = BS * LS;   // 4096
    const int Mdec = BS * LT;   // 2048

    // ---- inputs (setup_inputs() dict order) ----
    const int*   src     = (const int*)d_in[0];
    const int*   tgt     = (const int*)d_in[1];
    const int*   msk_src = (const int*)d_in[2];
    const int*   msk_tgt = (const int*)d_in[3];
    const float* src_emb = (const float*)d_in[4];
    const float* tgt_emb = (const float*)d_in[5];
    const float* ewq_w = (const float*)d_in[6],  *ewq_b = (const float*)d_in[7];
    const float* ewk_w = (const float*)d_in[8],  *ewk_b = (const float*)d_in[9];
    const float* ewv_w = (const float*)d_in[10], *ewv_b = (const float*)d_in[11];
    const float* ewo_w = (const float*)d_in[12], *ewo_b = (const float*)d_in[13];
    const float* eff1_w = (const float*)d_in[14], *eff1_b = (const float*)d_in[15];
    const float* eff2_w = (const float*)d_in[16], *eff2_b = (const float*)d_in[17];
    const float* eln1_g = (const float*)d_in[18], *eln1_b = (const float*)d_in[19];
    const float* eln2_g = (const float*)d_in[20], *eln2_b = (const float*)d_in[21];
    const float* dwq_w = (const float*)d_in[22], *dwq_b = (const float*)d_in[23];
    const float* dwk_w = (const float*)d_in[24], *dwk_b = (const float*)d_in[25];
    const float* dwv_w = (const float*)d_in[26], *dwv_b = (const float*)d_in[27];
    const float* dwo_w = (const float*)d_in[28], *dwo_b = (const float*)d_in[29];
    const float* dff1_w = (const float*)d_in[30], *dff1_b = (const float*)d_in[31];
    const float* dff2_w = (const float*)d_in[32], *dff2_b = (const float*)d_in[33];
    const float* dln1_g = (const float*)d_in[34], *dln1_b = (const float*)d_in[35];
    const float* dln2_g = (const float*)d_in[36], *dln2_b = (const float*)d_in[37];
    const float* dln3_g = (const float*)d_in[38], *dln3_b = (const float*)d_in[39];
    const float* encn_g = (const float*)d_in[40], *encn_b = (const float*)d_in[41];
    const float* decn_g = (const float*)d_in[42], *decn_b = (const float*)d_in[43];
    const float* gen_w  = (const float*)d_in[44], *gen_b  = (const float*)d_in[45];

    float* out = (float*)d_out;

    // ---- workspace layout (floats) ----
    float* W = (float*)d_ws;
    size_t o = 0;
    float* X    = W + o; o += (size_t)Menc * D;      // encoder activations
    float* XN   = W + o; o += (size_t)Menc * D;      // normalized activations
    float* Qb   = W + o; o += (size_t)Menc * D;
    float* Kb   = W + o; o += (size_t)Menc * D;
    float* Vb   = W + o; o += (size_t)Menc * D;
    float* Zb   = W + o; o += (size_t)Menc * D;      // attention output
    float* Zenc = W + o; o += (size_t)Menc * D;      // encoder memory
    float* H    = W + o; o += (size_t)Menc * F;      // FF hidden
    float* Y    = W + o; o += (size_t)Mdec * D;      // decoder activations
    float* YN   = W + o; o += (size_t)Mdec * D;

    auto gemm = [&](const float* A, const float* B, const float* bias,
                    const float* res, float* C, int M, int N, int K, int flags) {
        dim3 grid(N / 64, M / 128);
        gemm_bf16_wmma<<<grid, 256, 0, stream>>>(A, B, bias, res, C, M, N, K, flags);
    };

    // ============================= ENCODER =============================
    embed_pe<<<(Menc * D) / 256, 256, 0, stream>>>(src, src_emb, X, LS);

    for (int i = 0; i < NL; ++i) {
        const size_t wDD = (size_t)i * D * D, wD = (size_t)i * D;
        const size_t wDF = (size_t)i * D * F, wF = (size_t)i * F;

        lnorm512<<<Menc, 256, 0, stream>>>(X, eln1_g + wD, eln1_b + wD, XN);
        gemm(XN, ewq_w + wDD, ewq_b + wD, nullptr, Qb, Menc, D, D, 0);
        gemm(XN, ewk_w + wDD, ewk_b + wD, nullptr, Kb, Menc, D, D, 0);
        gemm(XN, ewv_w + wDD, ewv_b + wD, nullptr, Vb, Menc, D, D, 0);
        attn_soft<<<dim3(LS, 8, BS), 256, 0, stream>>>(Qb, Kb, Vb, msk_src, Zb,
                                                       LS, LS, LS, 0);
        gemm(Zb, ewo_w + wDD, ewo_b + wD, X, X, Menc, D, D, 2);      // + residual

        lnorm512<<<Menc, 256, 0, stream>>>(X, eln2_g + wD, eln2_b + wD, XN);
        gemm(XN, eff1_w + wDF, eff1_b + wF, nullptr, H, Menc, F, D, 1);  // ReLU
        gemm(H, eff2_w + wDF, eff2_b + wD, X, X, Menc, D, F, 2);         // + residual
    }
    lnorm512<<<Menc, 256, 0, stream>>>(X, encn_g, encn_b, Zenc);

    // ============================= DECODER =============================
    embed_pe<<<(Mdec * D) / 256, 256, 0, stream>>>(tgt, tgt_emb, Y, LT);

    for (int i = 0; i < NL; ++i) {
        const size_t wDD = (size_t)i * D * D, wD = (size_t)i * D;
        const size_t wDF = (size_t)i * D * F, wF = (size_t)i * F;

        // --- masked self-attention ---
        lnorm512<<<Mdec, 256, 0, stream>>>(Y, dln1_g + wD, dln1_b + wD, YN);
        gemm(YN, dwq_w + wDD, dwq_b + wD, nullptr, Qb, Mdec, D, D, 0);
        gemm(YN, dwk_w + wDD, dwk_b + wD, nullptr, Kb, Mdec, D, D, 0);
        gemm(YN, dwv_w + wDD, dwv_b + wD, nullptr, Vb, Mdec, D, D, 0);
        attn_soft<<<dim3(LT, 8, BS), 256, 0, stream>>>(Qb, Kb, Vb, msk_tgt, Zb,
                                                       LT, LT, LT * LT, LT);
        gemm(Zb, dwo_w + wDD, dwo_b + wD, Y, Y, Mdec, D, D, 2);

        // --- cross-attention (same shared weights, as in reference) ---
        lnorm512<<<Mdec, 256, 0, stream>>>(Y, dln2_g + wD, dln2_b + wD, YN);
        gemm(YN,   dwq_w + wDD, dwq_b + wD, nullptr, Qb, Mdec, D, D, 0);
        gemm(Zenc, dwk_w + wDD, dwk_b + wD, nullptr, Kb, Menc, D, D, 0);
        gemm(Zenc, dwv_w + wDD, dwv_b + wD, nullptr, Vb, Menc, D, D, 0);
        attn_soft<<<dim3(LT, 8, BS), 256, 0, stream>>>(Qb, Kb, Vb, msk_src, Zb,
                                                       LT, LS, LS, 0);
        gemm(Zb, dwo_w + wDD, dwo_b + wD, Y, Y, Mdec, D, D, 2);

        // --- feed-forward ---
        lnorm512<<<Mdec, 256, 0, stream>>>(Y, dln3_g + wD, dln3_b + wD, YN);
        gemm(YN, dff1_w + wDF, dff1_b + wF, nullptr, H, Mdec, F, D, 1);
        gemm(H, dff2_w + wDF, dff2_b + wD, Y, Y, Mdec, D, F, 2);
    }
    lnorm512<<<Mdec, 256, 0, stream>>>(Y, decn_g, decn_b, YN);

    // ============================= GENERATOR =============================
    gemm(YN, gen_w, gen_b, nullptr, out, Mdec, VT, D, 0);      // logits -> d_out
    log_softmax_rows<<<Mdec, 256, 0, stream>>>(out, VT);       // in-place
}
